// QGNN_50929722196863
// MI455X (gfx1250) — matrix operand order repaired and checked by reading, not compiled, not run on hardware
//
#include <hip/hip_runtime.h>
#include <hip/hip_bf16.h>
#include <math.h>

// ---------------------------------------------------------------------------
// QGNN message-passing network for MI455X (gfx1250, wave32).
// GEMMs (GRU gates + attention heads) run on v_wmma_f32_16x16x32_bf16.
// Everything is L2-resident (~35MB working set vs 192MB L2).
// ---------------------------------------------------------------------------

#define C        160
#define NUM_MSG  20
#define EPS      1e-5f
#define SLOPE    0.01f
#define RNUM     150
#define INUM     50

typedef __bf16 bf16_t;
typedef __attribute__((ext_vector_type(16))) __bf16 v16bf;
typedef __attribute__((ext_vector_type(8)))  float  v8f;

__device__ __forceinline__ float sigmoidf_(float x) { return 1.0f / (1.0f + __expf(-x)); }
__device__ __forceinline__ float leakyf_(float x)   { return x >= 0.0f ? x : SLOPE * x; }

__device__ __forceinline__ float warp_sum(float v) {
#pragma unroll
  for (int off = 16; off > 0; off >>= 1) v += __shfl_xor(v, off, 32);
  return v;
}

// ---- WMMA operand loaders (16-bit, wave32; contiguous k, same order in A and B) ---------------------
// A matrix 16x32 (MxK), row-major source, ld = row stride (elements):
//   lane L: row = rowBase + (L&15); khalf = (L<16)?0:16
//   elems 0..15 -> K = kBase + khalf + i
__device__ __forceinline__ v16bf load_A16x32(const bf16_t* __restrict__ M, int ld,
                                             int rowBase, int kBase, int lane) {
  const int r = rowBase + (lane & 15);
  const int khalf = (lane < 16) ? 0 : 16;
  const bf16_t* p = M + (size_t)r * ld + kBase;
  v16bf a;
#pragma unroll
  for (int i = 0; i < 16; ++i) a[i] = p[khalf + i];
  return a;
}

// B matrix 32x16 (KxN) where B = W^T and W is row-major [outCols x K]:
//   lane L: col n = colBase + (L&15); koff = (L<16)?0:16
//   elems 0..15 -> K = kBase + koff + i   (16 contiguous bf16 per lane)
__device__ __forceinline__ v16bf load_B_WT(const bf16_t* __restrict__ W, int ldw,
                                           int colBase, int kBase, int lane) {
  const int n = colBase + (lane & 15);
  const int koff = (lane < 16) ? 0 : 16;
  const bf16_t* p = W + (size_t)n * ldw + kBase + koff;
  v16bf b;
#pragma unroll
  for (int i = 0; i < 16; ++i) b[i] = p[i];
  return b;
}

__device__ __forceinline__ v8f wmma_bf16(v16bf a, v16bf b, v8f c) {
  // D = A*B + C, f32 accumulate
  return __builtin_amdgcn_wmma_f32_16x16x32_bf16(false, a, false, b, (short)0, c,
                                                 false, false);
}

// ---------------------------------------------------------------------------
// Kernels
// ---------------------------------------------------------------------------

// h = zero_right_pad(x, C); ew = 1
__global__ void k_init(const float* __restrict__ x, float* __restrict__ h,
                       bf16_t* __restrict__ hbf, float* __restrict__ ew,
                       int N, int E) {
  const int stride = gridDim.x * blockDim.x;
  const int total = N * C;
  for (int idx = blockIdx.x * blockDim.x + threadIdx.x; idx < total; idx += stride) {
    const int c = idx % C;
    const int n = idx / C;
    const float v = (c < 32) ? x[n * 32 + c] : 0.0f;
    h[idx] = v;
    hbf[idx] = (bf16_t)v;
  }
  for (int e = blockIdx.x * blockDim.x + threadIdx.x; e < E; e += stride)
    ew[e] = 1.0f;
}

__global__ void k_cvt_bf16(const float* __restrict__ in, bf16_t* __restrict__ out, int n) {
  const int stride = gridDim.x * blockDim.x;
  for (int i = blockIdx.x * blockDim.x + threadIdx.x; i < n; i += stride)
    out[i] = (bf16_t)in[i];
}

// agg[dst] += ew[e] * (h[src] + bias)   (agg pre-zeroed)
__global__ void k_aggregate(const float* __restrict__ h, const float* __restrict__ bias,
                            const float* __restrict__ ew, const int* __restrict__ ei,
                            float* __restrict__ agg, int E) {
  const int stride = gridDim.x * blockDim.x;
  const int total = E * C;
  for (int idx = blockIdx.x * blockDim.x + threadIdx.x; idx < total; idx += stride) {
    const int e = idx / C;
    const int c = idx - e * C;
    const int s = ei[e];
    const int d = ei[E + e];
    atomicAdd(&agg[d * C + c], ew[e] * (h[s * C + c] + bias[c]));
  }
}

// Fused GRU cell via WMMA. One wave computes a 16-node x 16-col tile of the
// gate space; r/z accumulate gi+gh into one accumulator each; inn and hn kept
// separate (n = tanh(inn + r*hn)). Writes pre-norm h_new.
__global__ void k_gru_gemm(const bf16_t* __restrict__ Mbf, const bf16_t* __restrict__ Hbf,
                           const bf16_t* __restrict__ Wih, const bf16_t* __restrict__ Whh,
                           const float* __restrict__ b_ih, const float* __restrict__ b_hh,
                           const float* __restrict__ Hf, float* __restrict__ Hpre, int N) {
  const int wave = (blockIdx.x * blockDim.x + threadIdx.x) >> 5;
  const int lane = threadIdx.x & 31;
  const int tilesPerRow = C / 16;                    // 10
  const int numTiles = (N / 16) * tilesPerRow;
  if (wave >= numTiles) return;                      // wave-uniform guard
  const int rowBase = (wave / tilesPerRow) * 16;
  const int colBase = (wave % tilesPerRow) * 16;

  v8f acc_r = {}, acc_z = {}, acc_n = {}, acc_hn = {};
#pragma unroll
  for (int kb = 0; kb < C; kb += 32) {
    const v16bf am = load_A16x32(Mbf, C, rowBase, kb, lane);
    const v16bf ah = load_A16x32(Hbf, C, rowBase, kb, lane);
    acc_r  = wmma_bf16(am, load_B_WT(Wih, C, colBase,         kb, lane), acc_r);
    acc_r  = wmma_bf16(ah, load_B_WT(Whh, C, colBase,         kb, lane), acc_r);
    acc_z  = wmma_bf16(am, load_B_WT(Wih, C, C + colBase,     kb, lane), acc_z);
    acc_z  = wmma_bf16(ah, load_B_WT(Whh, C, C + colBase,     kb, lane), acc_z);
    acc_n  = wmma_bf16(am, load_B_WT(Wih, C, 2 * C + colBase, kb, lane), acc_n);
    acc_hn = wmma_bf16(ah, load_B_WT(Whh, C, 2 * C + colBase, kb, lane), acc_hn);
  }

  const int col = colBase + (lane & 15);
  const float brr = b_ih[col] + b_hh[col];
  const float bzz = b_ih[C + col] + b_hh[C + col];
  const float bni = b_ih[2 * C + col];
  const float bnh = b_hh[2 * C + col];
  const int rowOff = (lane < 16) ? 0 : 8;            // D layout: M = v + rowOff
#pragma unroll
  for (int v = 0; v < 8; ++v) {
    const int node = rowBase + rowOff + v;
    const float r = sigmoidf_(acc_r[v] + brr);
    const float z = sigmoidf_(acc_z[v] + bzz);
    const float n = tanhf(acc_n[v] + bni + r * (acc_hn[v] + bnh));
    const float hold = Hf[node * C + col];
    Hpre[node * C + col] = (1.0f - z) * n + z * hold;
  }
}

// a_pre = h@WA^T + bA, b_pre = h@WB^T + bB  (one wave per 16x16 tile)
__global__ void k_ab_gemm(const bf16_t* __restrict__ Hbf,
                          const bf16_t* __restrict__ WAb, const bf16_t* __restrict__ WBb,
                          const float* __restrict__ bA, const float* __restrict__ bB,
                          float* __restrict__ Apre, float* __restrict__ Bpre, int N) {
  const int wave = (blockIdx.x * blockDim.x + threadIdx.x) >> 5;
  const int lane = threadIdx.x & 31;
  const int tilesPerRow = C / 16;
  const int numTiles = (N / 16) * tilesPerRow;
  if (wave >= numTiles) return;
  const int rowBase = (wave / tilesPerRow) * 16;
  const int colBase = (wave % tilesPerRow) * 16;

  v8f accA = {}, accB = {};
#pragma unroll
  for (int kb = 0; kb < C; kb += 32) {
    const v16bf ah = load_A16x32(Hbf, C, rowBase, kb, lane);
    accA = wmma_bf16(ah, load_B_WT(WAb, C, colBase, kb, lane), accA);
    accB = wmma_bf16(ah, load_B_WT(WBb, C, colBase, kb, lane), accB);
  }
  const int col = colBase + (lane & 15);
  const float ba = bA[col], bb = bB[col];
  const int rowOff = (lane < 16) ? 0 : 8;
#pragma unroll
  for (int v = 0; v < 8; ++v) {
    const int node = rowBase + rowOff + v;
    Apre[node * C + col] = accA[v] + ba;
    Bpre[node * C + col] = accB[v] + bb;
  }
}

// out = leaky(LayerNorm(in)) per row of C; one wave per row. Optional bf16 copy.
__global__ void k_ln_leaky(const float* __restrict__ in, const float* __restrict__ g,
                           const float* __restrict__ b, float* __restrict__ outf,
                           bf16_t* __restrict__ outbf, int rows) {
  const int row = (blockIdx.x * blockDim.x + threadIdx.x) >> 5;
  const int lane = threadIdx.x & 31;
  if (row >= rows) return;
  const float* p = in + (size_t)row * C;
  float vals[5];
  float s = 0.0f, s2 = 0.0f;
#pragma unroll
  for (int i = 0; i < 5; ++i) {
    const float v = p[lane + 32 * i];
    vals[i] = v; s += v; s2 += v * v;
  }
  s = warp_sum(s); s2 = warp_sum(s2);
  const float mu = s * (1.0f / C);
  const float var = s2 * (1.0f / C) - mu * mu;
  const float rstd = rsqrtf(var + EPS);
#pragma unroll
  for (int i = 0; i < 5; ++i) {
    const int c = lane + 32 * i;
    const float y = leakyf_(g[c] * (vals[i] - mu) * rstd + b[c]);
    outf[(size_t)row * C + c] = y;
    if (outbf) outbf[(size_t)row * C + c] = (bf16_t)y;
  }
}

// ew[e] = sigmoid(dot(a[src], b[dst])); one wave per edge
__global__ void k_edge_dot(const float* __restrict__ A, const float* __restrict__ B,
                           const int* __restrict__ ei, float* __restrict__ ew, int E) {
  const int e = (blockIdx.x * blockDim.x + threadIdx.x) >> 5;
  const int lane = threadIdx.x & 31;
  if (e >= E) return;
  const float* pa = A + (size_t)ei[e] * C;
  const float* pb = B + (size_t)ei[E + e] * C;
  float acc = 0.0f;
#pragma unroll
  for (int i = 0; i < 5; ++i) acc += pa[lane + 32 * i] * pb[lane + 32 * i];
  acc = warp_sum(acc);
  if (lane == 0) ew[e] = sigmoidf_(acc);
}

// Final head: q = leaky(LN(targets @ WQ^T + bQ)); o = sigmoid((q*h_iface)@WF^T + bF)
// One wave per output row (G*I = 2500 rows) — tiny, VALU only.
__global__ void k_head(const float* __restrict__ H, const float* __restrict__ targets,
                       const float* __restrict__ WQ, const float* __restrict__ bQ,
                       const float* __restrict__ lnQg, const float* __restrict__ lnQb,
                       const float* __restrict__ WF, const float* __restrict__ bF,
                       float* __restrict__ out, int G) {
  const int row = (blockIdx.x * blockDim.x + threadIdx.x) >> 5;
  const int lane = threadIdx.x & 31;
  const int total = G * INUM;
  if (row >= total) return;
  const int g = row / INUM;
  const int i = row - g * INUM;
  const int node = g * (RNUM + INUM) + RNUM + i;

  const float* tg = targets + g * 32;
  float t[5];
  float s = 0.0f, s2 = 0.0f;
#pragma unroll
  for (int j = 0; j < 5; ++j) {
    const int c = lane + 32 * j;
    float acc = bQ[c];
    const float* w = WQ + c * 32;
#pragma unroll
    for (int k = 0; k < 32; ++k) acc += tg[k] * w[k];
    t[j] = acc; s += acc; s2 += acc * acc;
  }
  s = warp_sum(s); s2 = warp_sum(s2);
  const float mu = s * (1.0f / C);
  const float var = s2 * (1.0f / C) - mu * mu;
  const float rstd = rsqrtf(var + EPS);

  const float* hrow = H + (size_t)node * C;
  float acc = 0.0f;
#pragma unroll
  for (int j = 0; j < 5; ++j) {
    const int c = lane + 32 * j;
    const float q = leakyf_(lnQg[c] * (t[j] - mu) * rstd + lnQb[c]);
    acc += q * hrow[c] * WF[c];
  }
  acc = warp_sum(acc);
  if (lane == 0) out[row] = sigmoidf_(acc + bF[0]);
}

// ---------------------------------------------------------------------------
extern "C" void kernel_launch(void* const* d_in, const int* in_sizes, int n_in,
                              void* d_out, int out_size, void* d_ws, size_t ws_size,
                              hipStream_t stream) {
  const float* x       = (const float*)d_in[0];
  const int*   ei      = (const int*)d_in[1];
  const float* targets = (const float*)d_in[4];
  const float* bias    = (const float*)d_in[5];
  const float* W_ih    = (const float*)d_in[6];
  const float* W_hh    = (const float*)d_in[7];
  const float* b_ih    = (const float*)d_in[8];
  const float* b_hh    = (const float*)d_in[9];
  const float* ln0g    = (const float*)d_in[10];
  const float* ln0b    = (const float*)d_in[11];
  const float* WA      = (const float*)d_in[12];
  const float* bA      = (const float*)d_in[13];
  const float* lnAg    = (const float*)d_in[14];
  const float* lnAb    = (const float*)d_in[15];
  const float* WB      = (const float*)d_in[16];
  const float* bB      = (const float*)d_in[17];
  const float* WQ      = (const float*)d_in[20];
  const float* bQ      = (const float*)d_in[21];
  const float* lnQg    = (const float*)d_in[22];
  const float* lnQb    = (const float*)d_in[23];
  const float* WF      = (const float*)d_in[24];
  const float* bF      = (const float*)d_in[25];

  const int N = in_sizes[0] / 32;     // 10000 (multiple of 16)
  const int E = in_sizes[1] / 2;      // 160000
  const int G = in_sizes[2];          // 50

  // Workspace carve (~36 MB total)
  char* w = (char*)d_ws;
  auto carve = [&](size_t bytes) -> char* {
    char* p = w; w += (bytes + 255) & ~(size_t)255; return p;
  };
  float*  h    = (float*)carve((size_t)N * C * 4);
  float*  hpre = (float*)carve((size_t)N * C * 4);
  float*  agg  = (float*)carve((size_t)N * C * 4);
  float*  Aab  = (float*)carve((size_t)N * C * 4);
  float*  Bab  = (float*)carve((size_t)N * C * 4);
  float*  ew   = (float*)carve((size_t)E * 4);
  bf16_t* hbf  = (bf16_t*)carve((size_t)N * C * 2);
  bf16_t* mbf  = (bf16_t*)carve((size_t)N * C * 2);
  bf16_t* Wihb = (bf16_t*)carve((size_t)3 * C * C * 2);
  bf16_t* Whhb = (bf16_t*)carve((size_t)3 * C * C * 2);
  bf16_t* WAb  = (bf16_t*)carve((size_t)C * C * 2);
  bf16_t* WBb  = (bf16_t*)carve((size_t)C * C * 2);

  // Weight bf16 conversion (deterministic, redone every call)
  k_cvt_bf16<<<(3 * C * C + 255) / 256, 256, 0, stream>>>(W_ih, Wihb, 3 * C * C);
  k_cvt_bf16<<<(3 * C * C + 255) / 256, 256, 0, stream>>>(W_hh, Whhb, 3 * C * C);
  k_cvt_bf16<<<(C * C + 255) / 256, 256, 0, stream>>>(WA, WAb, C * C);
  k_cvt_bf16<<<(C * C + 255) / 256, 256, 0, stream>>>(WB, WBb, C * C);

  k_init<<<4096, 256, 0, stream>>>(x, h, hbf, ew, N, E);

  const int numTiles   = (N / 16) * (C / 16);
  const int gemmBlocks = (numTiles + 7) / 8;             // 8 waves / block
  const int lnBlocks   = (N * 32 + 255) / 256;           // 1 wave / row
  const int edBlocks   = (E * 32 + 255) / 256;           // 1 wave / edge
  const int cvtBlocks  = (N * C + 255) / 256;

  for (int it = 0; it < NUM_MSG; ++it) {
    hipMemsetAsync(agg, 0, (size_t)N * C * 4, stream);
    k_aggregate<<<4096, 256, 0, stream>>>(h, bias, ew, ei, agg, E);
    k_cvt_bf16<<<cvtBlocks, 256, 0, stream>>>(agg, mbf, N * C);
    k_gru_gemm<<<gemmBlocks, 256, 0, stream>>>(mbf, hbf, Wihb, Whhb, b_ih, b_hh,
                                               h, hpre, N);
    k_ln_leaky<<<lnBlocks, 256, 0, stream>>>(hpre, ln0g, ln0b, h, hbf, N);
    k_ab_gemm<<<gemmBlocks, 256, 0, stream>>>(hbf, WAb, WBb, bA, bB, Aab, Bab, N);
    k_ln_leaky<<<lnBlocks, 256, 0, stream>>>(Aab, lnAg, lnAb, Aab, (bf16_t*)nullptr, N);
    k_ln_leaky<<<lnBlocks, 256, 0, stream>>>(Bab, (const float*)d_in[18],
                                             (const float*)d_in[19], Bab,
                                             (bf16_t*)nullptr, N);
    k_edge_dot<<<edBlocks, 256, 0, stream>>>(Aab, Bab, ei, ew, E);
  }

  k_head<<<(G * INUM * 32 + 255) / 256, 256, 0, stream>>>(
      h, targets, WQ, bQ, lnQg, lnQb, WF, bF, (float*)d_out, G);
}
